// FP8TransformerLayer_66606352826379
// MI455X (gfx1250) — 8-bit matrix operand order repaired and checked by reading, compiled, not run on hardware
//
#include <hip/hip_runtime.h>
#include <hip/hip_bf16.h>

typedef __attribute__((ext_vector_type(16))) int   v16i;
typedef __attribute__((ext_vector_type(8)))  float v8f;

#define QBS 128
#define E4M3_MAX 448.0f
#define E5M2_MAX 57344.0f

#if __has_builtin(__builtin_amdgcn_wmma_f32_16x16x128_bf8_bf8)
#define WMMA_BF8_128(a,b,c) __builtin_amdgcn_wmma_f32_16x16x128_bf8_bf8((a),(b),(short)0,(c),false,false)
#else
#define WMMA_BF8_128(a,b,c) __builtin_amdgcn_wmma_f32_16x16x128_fp8_fp8((a),(b),(short)0,(c),false,false)
#endif
#define WMMA_FP8_128(a,b,c) __builtin_amdgcn_wmma_f32_16x16x128_fp8_fp8((a),(b),(short)0,(c),false,false)

__device__ __forceinline__ unsigned char f32_to_e5m2(float f) {
  _Float16 h = (_Float16)f;
  unsigned short hb = __builtin_bit_cast(unsigned short, h);
  unsigned short lsb = (unsigned short)((hb >> 8) & 1u);
  hb = (unsigned short)(hb + 0x7Fu + lsb);
  return (unsigned char)(hb >> 8);
}

__device__ __forceinline__ unsigned char f32_to_e4m3(float f) {
  unsigned int u = __builtin_bit_cast(unsigned int, f);
  unsigned char sign = (unsigned char)((u >> 24) & 0x80u);
  float a = __builtin_fabsf(f);
  if (!(a > 0.0f)) return sign;
  if (a >= 448.0f) return (unsigned char)(sign | 0x7Eu);
  int ex = (int)((__builtin_bit_cast(unsigned int, a) >> 23) & 0xFF) - 127;
  if (ex < -6) ex = -6;
  float m = __builtin_ldexpf(a, 3 - ex);
  int mi = (int)__builtin_rintf(m);
  if (mi >= 16) { mi >>= 1; ex += 1; }
  unsigned int bits = (mi < 8) ? (unsigned)mi
                               : ((((unsigned)(ex + 7)) << 3) | (unsigned)(mi - 8));
  return (unsigned char)(sign | bits);
}

__device__ __forceinline__ float wred_max(float x) {
#pragma unroll
  for (int m = 16; m >= 1; m >>= 1) x = fmaxf(x, __shfl_xor(x, m, 32));
  return x;
}
__device__ __forceinline__ float wred_sum(float x) {
#pragma unroll
  for (int m = 16; m >= 1; m >>= 1) x += __shfl_xor(x, m, 32);
  return x;
}

__device__ __forceinline__ unsigned lds_addr(const void* p) {
  return (unsigned)(size_t)p;
}

template <int E5>
__global__ void k_wquant(const float* __restrict__ w, unsigned char* __restrict__ wq, int n) {
  int i = blockIdx.x * 256 + threadIdx.x;
  if (i < n) wq[i] = E5 ? f32_to_e5m2(w[i]) : f32_to_e4m3(w[i]);
}

template <int E5>
__global__ void k_actquant(const float* __restrict__ src, unsigned char* __restrict__ dq,
                           float* __restrict__ scales, int K, int nkb) {
  int row  = blockIdx.x;
  int wave = threadIdx.x >> 5, lane = threadIdx.x & 31;
  int kb   = blockIdx.y * 8 + wave;
  const float* p = src + (size_t)row * K + kb * QBS + lane * 4;
  float x0 = p[0], x1 = p[1], x2 = p[2], x3 = p[3];
  float amax = fmaxf(fmaxf(fabsf(x0), fabsf(x1)), fmaxf(fabsf(x2), fabsf(x3)));
  amax = wred_max(amax);
  const float fm = E5 ? E5M2_MAX : E4M3_MAX;
  float s = fmaxf(amax / fm, 1e-12f);
  float inv = 1.0f / s;
  unsigned int b0, b1, b2, b3;
  if (E5) {
    b0 = f32_to_e5m2(x0 * inv); b1 = f32_to_e5m2(x1 * inv);
    b2 = f32_to_e5m2(x2 * inv); b3 = f32_to_e5m2(x3 * inv);
  } else {
    b0 = f32_to_e4m3(x0 * inv); b1 = f32_to_e4m3(x1 * inv);
    b2 = f32_to_e4m3(x2 * inv); b3 = f32_to_e4m3(x3 * inv);
  }
  unsigned int packed = b0 | (b1 << 8) | (b2 << 16) | (b3 << 24);
  *(unsigned int*)(dq + (size_t)row * K + kb * QBS + lane * 4) = packed;
  if (lane == 0) scales[(size_t)row * nkb + kb] = s;
}

template <int E5>
__global__ void __launch_bounds__(256)
k_gemm_fp8(const unsigned char* __restrict__ Aq, const float* __restrict__ As,
           const unsigned char* __restrict__ Wq, const float* __restrict__ Ws,
           const float* __restrict__ bias, float* __restrict__ out,
           int M, int N, int K) {
  __shared__ __align__(16) unsigned char sA[2 * 64 * 128];
  __shared__ __align__(16) unsigned char sB[2 * 128 * 128];

  const int nkb  = K >> 7;
  const int tid  = threadIdx.x;
  const int wave = tid >> 5, lane = tid & 31;
  const int half = lane >> 4, l16 = lane & 15;
  const int wm = wave & 3, wn = wave >> 2;
  const int mBlock = blockIdx.x * 64;
  const int nBlock = blockIdx.y * 128;
  const int mBase = mBlock + wm * 16;
  const int nBase = nBlock + wn * 64;

  auto stage = [&](int buf, int kb) {
#pragma unroll
    for (int r = 0; r < 2; ++r) {
      int c = tid + r * 256;
      const unsigned char* g =
          Aq + (size_t)(mBlock + (c >> 3)) * K + (size_t)kb * 128 + ((c & 7) << 4);
      unsigned la = lds_addr(sA + buf * 8192 + c * 16);
      asm volatile("global_load_async_to_lds_b128 %0, %1, off"
                   :: "v"(la), "v"(g) : "memory");
    }
#pragma unroll
    for (int r = 0; r < 4; ++r) {
      int c = tid + r * 256;
      const unsigned char* g =
          Wq + (size_t)(nBlock + (c >> 3)) * K + (size_t)kb * 128 + ((c & 7) << 4);
      unsigned lb = lds_addr(sB + buf * 16384 + c * 16);
      asm volatile("global_load_async_to_lds_b128 %0, %1, off"
                   :: "v"(lb), "v"(g) : "memory");
    }
  };

  v8f acc[4];
#pragma unroll
  for (int t = 0; t < 4; ++t)
#pragma unroll
    for (int i = 0; i < 8; ++i) acc[t][i] = 0.0f;

  stage(0, 0);
  for (int kb = 0; kb < nkb; ++kb) {
    const int cur = kb & 1;
    if (kb + 1 < nkb) {
      stage(cur ^ 1, kb + 1);
      asm volatile("s_wait_asynccnt 0x6" ::: "memory");
    } else {
      asm volatile("s_wait_asynccnt 0x0" ::: "memory");
    }
    __syncthreads();

    const unsigned char* sAp = sA + cur * 8192 + (wm * 16 + l16) * 128 + half * 16;
    v16i a;
#pragma unroll
    for (int c = 0; c < 8; ++c) {
      uint2 q = *(const uint2*)(sAp + (c >> 1) * 32 + (c & 1) * 8);
      int vlo = c << 1;
      a[vlo] = (int)q.x;
      a[vlo + 1] = (int)q.y;
    }
    float sx[8];
#pragma unroll
    for (int i = 0; i < 8; ++i)
      sx[i] = As[(size_t)(mBase + i + (half << 3)) * nkb + kb];

#pragma unroll
    for (int t = 0; t < 4; ++t) {
      const unsigned char* sBp =
          sB + cur * 16384 + (wn * 64 + t * 16 + l16) * 128 + half * 16;
      v16i b;
#pragma unroll
      for (int g2 = 0; g2 < 4; ++g2) {
        uint4 q = *(const uint4*)(sBp + g2 * 32);
        b[4 * g2 + 0] = (int)q.x; b[4 * g2 + 1] = (int)q.y;
        b[4 * g2 + 2] = (int)q.z; b[4 * g2 + 3] = (int)q.w;
      }
      v8f z = {};
      v8f p = E5 ? WMMA_BF8_128(a, b, z) : WMMA_FP8_128(a, b, z);
#pragma unroll
      for (int i = 0; i < 8; ++i) acc[t][i] += sx[i] * p[i];
    }
    __syncthreads();
  }

#pragma unroll
  for (int t = 0; t < 4; ++t) {
    int n = nBase + t * 16 + l16;
    float wsv = Ws[n >> 7];
    float bv = bias ? bias[n] : 0.0f;
#pragma unroll
    for (int i = 0; i < 8; ++i) {
      int row = mBase + i + (half << 3);
      out[(size_t)row * N + n] = acc[t][i] * wsv + bv;
    }
  }
}

__global__ void __launch_bounds__(256)
k_attn(const float* __restrict__ q, const float* __restrict__ k,
       const float* __restrict__ v, const int* __restrict__ mask,
       float* __restrict__ ctx, int B, int H, int T, int HD) {
  int wave = threadIdx.x >> 5, lane = threadIdx.x & 31;
  int w  = blockIdx.x * 8 + wave;
  int tq = w % T;
  int bh = w / T;
  int h  = bh % H, b = bh / H;
  int C  = H * HD;
  size_t rowq = ((size_t)(b * T + tq) * C + h * HD) + lane * 4;
  float q0 = q[rowq], q1 = q[rowq + 1], q2 = q[rowq + 2], q3 = q[rowq + 3];
  const float sc = 1.0f / sqrtf((float)HD);
  const int* mrow = mask + ((size_t)b * T + tq) * T;
  float m = -3.0e38f, l = 0.0f;
  float a0 = 0.f, a1 = 0.f, a2 = 0.f, a3 = 0.f;
  for (int tk = 0; tk <= tq; ++tk) {
    if (mrow[tk] == 0) continue;
    size_t rk = ((size_t)(b * T + tk) * C + h * HD) + lane * 4;
    float d = q0 * k[rk] + q1 * k[rk + 1] + q2 * k[rk + 2] + q3 * k[rk + 3];
    d = wred_sum(d) * sc;
    float nm   = fmaxf(m, d);
    float corr = __expf(m - nm);
    float p    = __expf(d - nm);
    l = l * corr + p;
    a0 = a0 * corr + p * v[rk];
    a1 = a1 * corr + p * v[rk + 1];
    a2 = a2 * corr + p * v[rk + 2];
    a3 = a3 * corr + p * v[rk + 3];
    m = nm;
  }
  float inv = 1.0f / l;
  ctx[rowq] = a0 * inv; ctx[rowq + 1] = a1 * inv;
  ctx[rowq + 2] = a2 * inv; ctx[rowq + 3] = a3 * inv;
}

__global__ void k_gelu(float* __restrict__ h, int n) {
  int i = blockIdx.x * 256 + threadIdx.x;
  if (i < n) {
    float x = h[i];
    h[i] = 0.5f * x * (1.0f + erff(x * 0.70710678118654752f));
  }
}

__global__ void __launch_bounds__(256)
k_addln(const float* __restrict__ xa, const float* __restrict__ xb,
        const float* __restrict__ g, const float* __restrict__ gs,
        const float* __restrict__ bb, const float* __restrict__ bsc,
        float* __restrict__ out, int C) {
  int wave = threadIdx.x >> 5, lane = threadIdx.x & 31;
  int row = blockIdx.x * 8 + wave;
  const float* pa = xa + (size_t)row * C;
  const float* pb = xb + (size_t)row * C;
  float s = 0.f, ss = 0.f;
  for (int i = lane; i < C; i += 32) {
    float t = pa[i] + pb[i];
    s += t; ss += t * t;
  }
  s = wred_sum(s); ss = wred_sum(ss);
  float mean = s / (float)C;
  float var  = ss / (float)C - mean * mean;
  float inv  = rsqrtf(var + 1e-5f);
  float* po = out + (size_t)row * C;
  for (int i = lane; i < C; i += 32) {
    float t = pa[i] + pb[i];
    po[i] = (t - mean) * inv * (g[i] * gs[i >> 7]) + bb[i] * bsc[i >> 7];
  }
}

static inline unsigned char* carve(unsigned char*& p, size_t bytes) {
  unsigned char* r = p;
  p += (bytes + 255) & ~(size_t)255;
  return r;
}

extern "C" void kernel_launch(void* const* d_in, const int* in_sizes, int n_in,
                              void* d_out, int out_size, void* d_ws, size_t ws_size,
                              hipStream_t stream) {
  (void)in_sizes; (void)n_in; (void)out_size; (void)ws_size;
  const int B = 2, T = 1024, C = 2048, MLP = 8192, H = 16, HD = 128;
  const int M = B * T;
  const int NKB_C = C / QBS;
  const int NKB_M = MLP / QBS;

  const float* x    = (const float*)d_in[0];
  const int*   mask = (const int*)d_in[1];
  const float* wq_w = (const float*)d_in[2];  const float* qs   = (const float*)d_in[3];
  const float* wk_w = (const float*)d_in[4];  const float* ks   = (const float*)d_in[5];
  const float* wv_w = (const float*)d_in[6];  const float* vs   = (const float*)d_in[7];
  const float* wo_w = (const float*)d_in[8];  const float* os_  = (const float*)d_in[9];
  const float* fc1w = (const float*)d_in[10]; const float* fc1s = (const float*)d_in[11];
  const float* fc1b = (const float*)d_in[12];
  const float* fc2w = (const float*)d_in[13]; const float* fc2s = (const float*)d_in[14];
  const float* fc2b = (const float*)d_in[15];
  const float* l1g = (const float*)d_in[16]; const float* l1gs = (const float*)d_in[17];
  const float* l1b = (const float*)d_in[18]; const float* l1bs = (const float*)d_in[19];
  const float* l2g = (const float*)d_in[20]; const float* l2gs = (const float*)d_in[21];
  const float* l2b = (const float*)d_in[22]; const float* l2bs = (const float*)d_in[23];
  float* outp = (float*)d_out;

  unsigned char* wsp = (unsigned char*)d_ws;
  unsigned char* wqq = carve(wsp, (size_t)C * C);
  unsigned char* wkq = carve(wsp, (size_t)C * C);
  unsigned char* wvq = carve(wsp, (size_t)C * C);
  unsigned char* woq = carve(wsp, (size_t)C * C);
  unsigned char* f1q = carve(wsp, (size_t)MLP * C);
  unsigned char* f2q = carve(wsp, (size_t)C * MLP);
  unsigned char* xq  = carve(wsp, (size_t)M * C);
  float* sx  = (float*)carve(wsp, (size_t)M * NKB_C * 4);
  float* qb  = (float*)carve(wsp, (size_t)M * C * 4);
  float* kbuf= (float*)carve(wsp, (size_t)M * C * 4);
  float* vb  = (float*)carve(wsp, (size_t)M * C * 4);
  float* ctx = (float*)carve(wsp, (size_t)M * C * 4);
  unsigned char* cq = carve(wsp, (size_t)M * C);
  float* scx = (float*)carve(wsp, (size_t)M * NKB_C * 4);
  float* ao  = (float*)carve(wsp, (size_t)M * C * 4);
  float* x1  = (float*)carve(wsp, (size_t)M * C * 4);
  unsigned char* x1q = carve(wsp, (size_t)M * C);
  float* s1  = (float*)carve(wsp, (size_t)M * NKB_C * 4);
  float* hb  = (float*)carve(wsp, (size_t)M * MLP * 4);
  unsigned char* hq = carve(wsp, (size_t)M * MLP);
  float* sh  = (float*)carve(wsp, (size_t)M * NKB_M * 4);
  float* ffn = (float*)carve(wsp, (size_t)M * C * 4);

  {
    int n = C * C, g = (n + 255) / 256;
    k_wquant<1><<<g, 256, 0, stream>>>(wq_w, wqq, n);
    k_wquant<1><<<g, 256, 0, stream>>>(wk_w, wkq, n);
    k_wquant<1><<<g, 256, 0, stream>>>(wv_w, wvq, n);
    k_wquant<1><<<g, 256, 0, stream>>>(wo_w, woq, n);
    int n2 = MLP * C, g2 = (n2 + 255) / 256;
    k_wquant<0><<<g2, 256, 0, stream>>>(fc1w, f1q, n2);
    k_wquant<0><<<g2, 256, 0, stream>>>(fc2w, f2q, n2);
  }

  k_actquant<1><<<dim3(M, NKB_C / 8), 256, 0, stream>>>(x, xq, sx, C, NKB_C);

  dim3 gC(M / 64, C / 128);
  k_gemm_fp8<1><<<gC, 256, 0, stream>>>(xq, sx, wqq, qs, nullptr, qb, M, C, C);
  k_gemm_fp8<1><<<gC, 256, 0, stream>>>(xq, sx, wkq, ks, nullptr, kbuf, M, C, C);
  k_gemm_fp8<1><<<gC, 256, 0, stream>>>(xq, sx, wvq, vs, nullptr, vb, M, C, C);

  k_attn<<<(B * H * T) / 8, 256, 0, stream>>>(qb, kbuf, vb, mask, ctx, B, H, T, HD);

  k_actquant<1><<<dim3(M, NKB_C / 8), 256, 0, stream>>>(ctx, cq, scx, C, NKB_C);
  k_gemm_fp8<1><<<gC, 256, 0, stream>>>(cq, scx, woq, os_, nullptr, ao, M, C, C);

  k_addln<<<M / 8, 256, 0, stream>>>(x, ao, l1g, l1gs, l1b, l1bs, x1, C);

  k_actquant<0><<<dim3(M, NKB_C / 8), 256, 0, stream>>>(x1, x1q, s1, C, NKB_C);
  k_gemm_fp8<0><<<dim3(M / 64, MLP / 128), 256, 0, stream>>>(x1q, s1, f1q, fc1s, fc1b, hb, M, MLP, C);
  k_gelu<<<(M * MLP) / 256, 256, 0, stream>>>(hb, M * MLP);

  k_actquant<0><<<dim3(M, NKB_M / 8), 256, 0, stream>>>(hb, hq, sh, MLP, NKB_M);
  k_gemm_fp8<0><<<gC, 256, 0, stream>>>(hq, sh, f2q, fc2s, fc2b, ffn, M, C, MLP);

  k_addln<<<M / 8, 256, 0, stream>>>(x1, ffn, l2g, l2gs, l2b, l2bs, outp, C);
}
